// Decoder_29549374997026
// MI455X (gfx1250) — matrix operand order repaired and checked by reading, not compiled, not run on hardware
//
#include <hip/hip_runtime.h>
#include <hip/hip_bf16.h>
#include <math.h>

// Problem constants (match reference)
#define B_  32
#define TD_ 64
#define TE_ 128
#define V_  32000
#define E_  256
#define H_  512

typedef __attribute__((ext_vector_type(16))) _Float16 v16h;
typedef __attribute__((ext_vector_type(8)))  float    v8f;

struct __align__(16) U4 { unsigned a, b, c, d; };
union AF { v16h h; unsigned u[8]; };

// ---------------------------------------------------------------------------
// One-time kernels
// ---------------------------------------------------------------------------

// K = einsum('bth,gh->btg', enc, Wk): one thread per output element
__global__ void enck_kernel(const float* __restrict__ enc,
                            const float* __restrict__ Wk,
                            float* __restrict__ Kp) {
  int idx = blockIdx.x * 256 + threadIdx.x;            // (B*TE)*H outputs
  int row = idx >> 9;                                   // b*TE + te
  int g   = idx & 511;
  const float4* e4 = (const float4*)(enc + (size_t)row * H_);
  const float4* w4 = (const float4*)(Wk  + (size_t)g   * H_);
  float s = 0.f;
  for (int j = 0; j < H_ / 4; ++j) {
    float4 a = e4[j], b = w4[j];
    s += a.x * b.x + a.y * b.y + a.z * b.z + a.w * b.w;
  }
  Kp[idx] = s;
}

// Xemb[b][t][e] = emb[X[b][t]][e]
__global__ void embed_kernel(const int* __restrict__ X,
                             const float* __restrict__ emb,
                             float* __restrict__ Xemb) {
  int idx = blockIdx.x * 256 + threadIdx.x;            // B*TD*E
  int e  = idx & (E_ - 1);
  int bt = idx >> 8;
  Xemb[idx] = emb[(size_t)X[bt] * E_ + e];
}

// fp32 -> f16 convert
__global__ void half_kernel(const float* __restrict__ src,
                            _Float16* __restrict__ dst, int n) {
  int i = blockIdx.x * 256 + threadIdx.x;
  if (i < n) dst[i] = (_Float16)src[i];
}

// split h0 (L,B,H) into layer buffers
__global__ void inith_kernel(const float* __restrict__ h0in,
                             float* __restrict__ h0, float* __restrict__ h1) {
  int i = blockIdx.x * 256 + threadIdx.x;
  if (i < B_ * H_) { h0[i] = h0in[i]; h1[i] = h0in[B_ * H_ + i]; }
}

// final hN (L,B,H) to tail of d_out
__global__ void hn_kernel(const float* __restrict__ h0,
                          const float* __restrict__ h1,
                          float* __restrict__ out) {
  int i = blockIdx.x * 256 + threadIdx.x;
  if (i < B_ * H_) { out[i] = h0[i]; out[B_ * H_ + i] = h1[i]; }
}

// ---------------------------------------------------------------------------
// Per-step: Bahdanau attention. One block per batch row b, 128 threads.
// Writes context into inp[b][0:512] and x_t into inp[b][512:768].
// ---------------------------------------------------------------------------
__global__ void attn_kernel(const float* __restrict__ h1,
                            const float* __restrict__ Wq,
                            const float* __restrict__ Kp,
                            const float* __restrict__ Wv,
                            const float* __restrict__ enc,
                            const unsigned char* __restrict__ mask,
                            const float* __restrict__ Xemb,
                            int t,
                            float* __restrict__ inp) {
  const int b = blockIdx.x, tid = threadIdx.x;         // 128 threads
  __shared__ float hS[H_], qS[H_], sc[TE_], red[TE_], stat[2];

  for (int i = tid; i < H_; i += 128) hS[i] = h1[b * H_ + i];
  __syncthreads();

  // q = h1[b] @ Wq^T
  for (int j = tid; j < H_; j += 128) {
    const float4* w  = (const float4*)(Wq + (size_t)j * H_);
    const float4* h4 = (const float4*)hS;
    float s = 0.f;
    for (int k = 0; k < H_ / 4; ++k) {
      float4 a = h4[k], bb = w[k];
      s += a.x * bb.x + a.y * bb.y + a.z * bb.z + a.w * bb.w;
    }
    qS[j] = s;
  }
  __syncthreads();

  // scores[te] = v . tanh(q + K[b,te,:])
  {
    const int te = tid;
    const float* Krow = Kp + ((size_t)b * TE_ + te) * H_;
    float s = 0.f;
    for (int h = 0; h < H_; ++h) s += Wv[h] * tanhf(qS[h] + Krow[h]);
    if (mask[b * TE_ + te]) s = -INFINITY;
    sc[te] = s;
  }
  __syncthreads();

  // softmax over TE=128
  red[tid] = sc[tid]; __syncthreads();
  for (int off = 64; off > 0; off >>= 1) {
    if (tid < off) red[tid] = fmaxf(red[tid], red[tid + off]);
    __syncthreads();
  }
  if (tid == 0) stat[0] = red[0];
  __syncthreads();
  float e = __expf(sc[tid] - stat[0]);
  red[tid] = e; __syncthreads();
  for (int off = 64; off > 0; off >>= 1) {
    if (tid < off) red[tid] += red[tid + off];
    __syncthreads();
  }
  if (tid == 0) stat[1] = red[0];
  __syncthreads();
  sc[tid] = e / stat[1];
  __syncthreads();

  // context = attn @ enc
  for (int h = tid; h < H_; h += 128) {
    const float* ebase = enc + (size_t)b * TE_ * H_ + h;
    float s = 0.f;
    for (int te = 0; te < TE_; ++te) s += sc[te] * ebase[te * H_];
    inp[b * (H_ + E_) + h] = s;
  }
  // x_t slice
  for (int e2 = tid; e2 < E_; e2 += 128)
    inp[b * (H_ + E_) + H_ + e2] = Xemb[((size_t)b * TD_ + t) * E_ + e2];
}

// ---------------------------------------------------------------------------
// Per-step: fused GRU cell, one thread per (b,i). Optionally emits f16 copy
// of the new hidden (layer 1 -> A-matrix rows for the WMMA projection).
// ---------------------------------------------------------------------------
__global__ void gru_kernel(const float* __restrict__ x, int xdim,
                           const float* __restrict__ hprev,
                           const float* __restrict__ Wih,
                           const float* __restrict__ Whh,
                           const float* __restrict__ bih,
                           const float* __restrict__ bhh,
                           float* __restrict__ hnew,
                           _Float16* __restrict__ out16) {
  const int g = blockIdx.x * blockDim.x + threadIdx.x; // B*H threads
  const int b = g >> 9, i = g & 511;
  const float4* xr = (const float4*)(x + (size_t)b * xdim);
  const float4* hr = (const float4*)(hprev + (size_t)b * H_);
  const int x4 = xdim >> 2;
  float gi[3], gh[3];
  for (int k = 0; k < 3; ++k) {
    const int row = k * H_ + i;
    const float4* w = (const float4*)(Wih + (size_t)row * xdim);
    float s = bih[row];
    for (int j = 0; j < x4; ++j) {
      float4 a = xr[j], bb = w[j];
      s += a.x * bb.x + a.y * bb.y + a.z * bb.z + a.w * bb.w;
    }
    gi[k] = s;
    const float4* wh = (const float4*)(Whh + (size_t)row * H_);
    float sh = bhh[row];
    for (int j = 0; j < H_ / 4; ++j) {
      float4 a = hr[j], bb = wh[j];
      sh += a.x * bb.x + a.y * bb.y + a.z * bb.z + a.w * bb.w;
    }
    gh[k] = sh;
  }
  float r  = 1.f / (1.f + __expf(-(gi[0] + gh[0])));
  float z  = 1.f / (1.f + __expf(-(gi[1] + gh[1])));
  float n  = tanhf(gi[2] + r * gh[2]);
  float hp = hprev[(size_t)b * H_ + i];
  float hn = (1.f - z) * n + z * hp;
  hnew[g] = hn;
  if (out16) out16[(size_t)b * H_ + i] = (_Float16)hn;
}

// ---------------------------------------------------------------------------
// Fused projection + log-softmax. 128 blocks x 256 threads (8 waves).
// Block owns 16 rows of A (GRU outputs, f16, rows r = t*B + b).
// Wave w sweeps 16-col vocab tiles n0 = 16w + 128j with WMMA f16->f32.
// ---------------------------------------------------------------------------
__global__ void __launch_bounds__(256)
proj_kernel(const _Float16* __restrict__ A16,      // [TD*B][H]
            const _Float16* __restrict__ W16,      // [V][H]  (Wd in f16)
            const float* __restrict__ bd,          // [V]
            float* __restrict__ out) {             // logp (B,TD,V)
  const int tid  = threadIdx.x;
  const int lane = tid & 31, wave = tid >> 5;
  const int r0   = blockIdx.x * 16;

  __shared__ _Float16 As[16 * H_];
  __shared__ float smax[16][8];
  __shared__ float rowmax[16];
  __shared__ float rowlogz[16];
  __shared__ float red[256];

  // Stage A tile (16 rows x 512 f16 = 16 KB) into LDS, shared by all waves
  {
    const U4* src = (const U4*)(A16 + (size_t)r0 * H_);
    U4* dst = (U4*)As;
    for (int i = tid; i < 16 * H_ / 8; i += 256) dst[i] = src[i];
  }
  __syncthreads();

  const int m  = lane & 15;          // N within tile (B-cols) / M row (A)
  const int hi = lane >> 4;          // half-wave select
  const int kselA = hi * 8;          // A lanes 16-31 start at K=8
  const int kselB = hi * 8;          // B lanes 16-31 start at K=8

  // d_out row bases: row r -> (b*TD + t)*V with b=r%B, t=r/B
  int rbase[8];
#pragma unroll
  for (int i = 0; i < 8; ++i) {
    int row = r0 + i + hi * 8;
    rbase[i] = ((row & (B_ - 1)) * TD_ + (row >> 5)) * V_;
  }
  float rmax[8];
#pragma unroll
  for (int i = 0; i < 8; ++i) rmax[i] = -INFINITY;

  const unsigned* As32 = (const unsigned*)As;

  // Pass 1: logits = A @ Wd^T + bd, track per-row max
  for (int n0 = wave * 16; n0 < V_; n0 += 128) {
    const int n = n0 + m;
    const float bias = bd[n];
    const _Float16* brow = W16 + (size_t)n * H_;
    if (n + 128 < V_) __builtin_prefetch(W16 + (size_t)(n + 128) * H_, 0, 0);
    v8f c = {};
    for (int kc = 0; kc < H_; kc += 32) {
      AF a, bb;
      const unsigned* ap = As32 + ((m * H_ + kc + kselA) >> 1);
      a.u[0] = ap[0];  a.u[1] = ap[1];  a.u[2] = ap[2];  a.u[3] = ap[3];
      a.u[4] = ap[8];  a.u[5] = ap[9];  a.u[6] = ap[10]; a.u[7] = ap[11];
      const U4* bp = (const U4*)(brow + kc + kselB);
      U4 b0 = bp[0], b1 = bp[2];
      bb.u[0] = b0.a; bb.u[1] = b0.b; bb.u[2] = b0.c; bb.u[3] = b0.d;
      bb.u[4] = b1.a; bb.u[5] = b1.b; bb.u[6] = b1.c; bb.u[7] = b1.d;
      c = __builtin_amdgcn_wmma_f32_16x16x32_f16(false, a.h, false, bb.h,
                                                 (short)0, c, false, false);
    }
#pragma unroll
    for (int i = 0; i < 8; ++i) {
      float x = c[i] + bias;
      out[rbase[i] + n] = x;
      rmax[i] = fmaxf(rmax[i], x);
    }
  }

  // Reduce row max across the 16 lanes of each half-wave, then across waves
#pragma unroll
  for (int i = 0; i < 8; ++i) {
    float v = rmax[i];
    for (int d = 1; d < 16; d <<= 1) v = fmaxf(v, __shfl_xor(v, d, 32));
    rmax[i] = v;
  }
  if (m == 0)
    for (int i = 0; i < 8; ++i) smax[i + hi * 8][wave] = rmax[i];
  __threadfence();
  __syncthreads();
  if (tid < 16) {
    float v = smax[tid][0];
    for (int w = 1; w < 8; ++w) v = fmaxf(v, smax[tid][w]);
    rowmax[tid] = v;
  }
  __syncthreads();

  // Pass 2: sum of exp per row (block's slab is L2-resident)
  for (int rr = 0; rr < 16; ++rr) {
    int row = r0 + rr;
    int base = ((row & (B_ - 1)) * TD_ + (row >> 5)) * V_;
    const float mx = rowmax[rr];
    const float4* p4 = (const float4*)(out + base);
    float s = 0.f;
    for (int c4 = tid; c4 < V_ / 4; c4 += 256) {
      float4 x = p4[c4];
      s += __expf(x.x - mx) + __expf(x.y - mx) +
           __expf(x.z - mx) + __expf(x.w - mx);
    }
    red[tid] = s; __syncthreads();
    for (int off = 128; off > 0; off >>= 1) {
      if (tid < off) red[tid] += red[tid + off];
      __syncthreads();
    }
    if (tid == 0) rowlogz[rr] = mx + __logf(red[0]);
    __syncthreads();
  }

  // Pass 3: logp = x - logZ
  for (int rr = 0; rr < 16; ++rr) {
    int row = r0 + rr;
    int base = ((row & (B_ - 1)) * TD_ + (row >> 5)) * V_;
    const float lz = rowlogz[rr];
    float4* p4 = (float4*)(out + base);
    for (int c4 = tid; c4 < V_ / 4; c4 += 256) {
      float4 x = p4[c4];
      x.x -= lz; x.y -= lz; x.z -= lz; x.w -= lz;
      p4[c4] = x;
    }
  }
}

// ---------------------------------------------------------------------------
// Host driver
// ---------------------------------------------------------------------------
extern "C" void kernel_launch(void* const* d_in, const int* in_sizes, int n_in,
                              void* d_out, int out_size, void* d_ws, size_t ws_size,
                              hipStream_t stream) {
  (void)in_sizes; (void)n_in; (void)out_size; (void)ws_size;
  const int*           X    = (const int*)d_in[0];
  const float*         enc  = (const float*)d_in[1];
  const float*         h0in = (const float*)d_in[2];
  const unsigned char* mask = (const unsigned char*)d_in[3];
  const float*         emb  = (const float*)d_in[4];
  const float*         Wq   = (const float*)d_in[5];
  const float*         Wk   = (const float*)d_in[6];
  const float*         Wv   = (const float*)d_in[7];
  const float*         Wih0 = (const float*)d_in[8];
  const float*         Whh0 = (const float*)d_in[9];
  const float*         bih0 = (const float*)d_in[10];
  const float*         bhh0 = (const float*)d_in[11];
  const float*         Wih1 = (const float*)d_in[12];
  const float*         Whh1 = (const float*)d_in[13];
  const float*         bih1 = (const float*)d_in[14];
  const float*         bhh1 = (const float*)d_in[15];
  const float*         Wd   = (const float*)d_in[16];
  const float*         bd   = (const float*)d_in[17];
  float* out = (float*)d_out;

  // Carve workspace
  char* p = (char*)d_ws;
  auto carve = [&](size_t bytes) -> void* {
    void* r = (void*)p;
    p += (bytes + 255) & ~(size_t)255;
    return r;
  };
  float*    Kp     = (float*)carve((size_t)B_ * TE_ * H_ * 4);        // 8 MB
  float*    Xemb   = (float*)carve((size_t)B_ * TD_ * E_ * 4);        // 2 MB
  _Float16* Wd16   = (_Float16*)carve((size_t)V_ * H_ * 2);           // 32 MB
  _Float16* outs16 = (_Float16*)carve((size_t)TD_ * B_ * H_ * 2);     // 2 MB
  float*    inp    = (float*)carve((size_t)B_ * (H_ + E_) * 4);
  float*    h0a    = (float*)carve((size_t)B_ * H_ * 4);
  float*    h0b    = (float*)carve((size_t)B_ * H_ * 4);
  float*    h1a    = (float*)carve((size_t)B_ * H_ * 4);
  float*    h1b    = (float*)carve((size_t)B_ * H_ * 4);

  // One-time setup
  enck_kernel<<<(B_ * TE_ * H_) / 256, 256, 0, stream>>>(enc, Wk, Kp);
  embed_kernel<<<(B_ * TD_ * E_) / 256, 256, 0, stream>>>(X, emb, Xemb);
  half_kernel<<<(V_ * H_ + 255) / 256, 256, 0, stream>>>(Wd, Wd16, V_ * H_);
  inith_kernel<<<(B_ * H_ + 255) / 256, 256, 0, stream>>>(h0in, h0a, h1a);

  // Recurrent decode loop
  float *h0c = h0a, *h0n = h0b, *h1c = h1a, *h1n = h1b;
  for (int t = 0; t < TD_; ++t) {
    attn_kernel<<<B_, 128, 0, stream>>>(h1c, Wq, Kp, Wv, enc, mask, Xemb, t, inp);
    gru_kernel<<<(B_ * H_) / 256, 256, 0, stream>>>(
        inp, H_ + E_, h0c, Wih0, Whh0, bih0, bhh0, h0n, (_Float16*)nullptr);
    gru_kernel<<<(B_ * H_) / 256, 256, 0, stream>>>(
        h0n, H_, h1c, Wih1, Whh1, bih1, bhh1, h1n,
        outs16 + (size_t)t * B_ * H_);
    float* tmp;
    tmp = h0c; h0c = h0n; h0n = tmp;
    tmp = h1c; h1c = h1n; h1n = tmp;
  }

  // Fused vocab projection + log-softmax (the WMMA heavy hitter)
  proj_kernel<<<(TD_ * B_) / 16, 256, 0, stream>>>(outs16, Wd16, bd, out);

  // hN tail
  hn_kernel<<<(B_ * H_ + 255) / 256, 256, 0, stream>>>(
      h0c, h1c, out + (size_t)B_ * TD_ * V_);
}
